// RIM_64613488001361
// MI455X (gfx1250) — matrix operand order repaired and checked by reading, not compiled, not run on hardware
//
#include <hip/hip_runtime.h>
#include <hip/hip_bf16.h>

// Problem constants (match reference)
#define Bsz 4096
#define Uu  6
#define Kk  4
#define Dd  512
#define Hh  512
#define HIN_ 2
#define QS_ 64
#define KS_ 64
#define VS_ 512
#define HC_ 4
#define QC_ 32
#define KC_ 32
#define VC_ 512
#define G3H 1536   // 3*H

typedef __attribute__((ext_vector_type(16))) __bf16 v16bf;
typedef __attribute__((ext_vector_type(8)))  float  v8f;

union Frag { unsigned int u[8]; v16bf v; };

__device__ __forceinline__ unsigned short f2bf(float f) {
  unsigned int u = __float_as_uint(f);
  u += 0x7FFFu + ((u >> 16) & 1u);           // round-to-nearest-even
  return (unsigned short)(u >> 16);
}

// ---------------- conversion / prep kernels ----------------

__global__ __launch_bounds__(256) void k_cvt(const float* __restrict__ s,
                                             unsigned short* __restrict__ d, int n) {
  int i = blockIdx.x * 256 + threadIdx.x;
  if (i < n) d[i] = f2bf(s[i]);
}

// src[z][R][C] f32 -> dst[z][C][R] bf16
__global__ __launch_bounds__(256) void k_transpose(const float* __restrict__ s,
                                                   unsigned short* __restrict__ d,
                                                   int R, int C) {
  int i = blockIdx.x * 256 + threadIdx.x;
  int z = blockIdx.z;
  if (i >= R * C) return;
  int r = i / C, c = i % C;
  d[(size_t)z * R * C + (size_t)c * R + r] = f2bf(s[(size_t)z * R * C + i]);
}

// W[z][R][NH*S] f32 -> Wm[z][R][S] bf16 = mean over NH head blocks; bias likewise
__global__ __launch_bounds__(256) void k_headmean(const float* __restrict__ W,
                                                  const float* __restrict__ b,
                                                  unsigned short* __restrict__ Wm,
                                                  float* __restrict__ bm,
                                                  int R, int NH, int S) {
  int i = blockIdx.x * 256 + threadIdx.x;
  int z = blockIdx.z;
  if (i >= R * S) return;
  int r = i / S, s = i % S;
  const float inv = 1.0f / (float)NH;
  float acc = 0.f;
  for (int h = 0; h < NH; ++h)
    acc += W[(size_t)z * R * NH * S + (size_t)r * NH * S + h * S + s];
  Wm[(size_t)z * R * S + i] = f2bf(acc * inv);
  if (r == 0) {
    float ba = 0.f;
    for (int h = 0; h < NH; ++h) ba += b[(size_t)z * NH * S + h * S + s];
    bm[(size_t)z * S + s] = ba * inv;
  }
}

// ---------------- WMMA bf16 GEMM: C[M,N] = A[M,K] @ B[K,N] + bias ----------------
// M%128==0, N%64==0, K%32==0. Batched via blockIdx.z with element strides.
// 8 waves, block tile 128x64, wave tile 32x32 (4 WMMA / K-step, 2x fragment reuse),
// double-buffered LDS; A staged via CDNA5 GLOBAL_LOAD_ASYNC_TO_LDS_B128 (ASYNCcnt).
#define BM 128
#define BN 64
#define BK 32

__global__ __launch_bounds__(256) void k_gemm(
    const unsigned short* __restrict__ A, long long strideA,
    const unsigned short* __restrict__ Bm, long long strideB,
    const float* __restrict__ bias, long long strideBias,
    float* __restrict__ C, long long strideC,
    int M, int N, int K) {
  __shared__ __align__(16) unsigned short lA[2][BM * BK];  // [m][k]
  __shared__ __align__(16) unsigned short lB[2][BN * BK];  // transposed: [n][k]

  const int z = blockIdx.z;
  A  += (size_t)z * strideA;
  Bm += (size_t)z * strideB;
  C  += (size_t)z * strideC;
  if (bias) bias += (size_t)z * strideBias;

  const int n0 = blockIdx.x * BN, m0 = blockIdx.y * BM;
  const int tid  = threadIdx.x;
  const int lane = tid & 31, w = tid >> 5;
  const int wm = w & 3, wn = w >> 2;         // 4 wave-rows x 2 wave-cols
  const int half = lane >> 4, l16 = lane & 15;

  v8f acc00 = {0.f, 0.f, 0.f, 0.f, 0.f, 0.f, 0.f, 0.f};
  v8f acc01 = acc00, acc10 = acc00, acc11 = acc00;

  const int ar = tid >> 2, akq = (tid & 3) * 8;  // A loader: rows ar, ar+64; 8 bf16 chunks
  const int bk = tid >> 3, bn = (tid & 7) * 8;   // B loader: 32 rows x 64 cols

  // issue async global->LDS copies for the A tile (two b128 per thread)
  auto issueA = [&](int kt, int buf) {
    unsigned long long g0 =
        (unsigned long long)(A + (size_t)(m0 + ar) * K + kt + akq);
    unsigned lo0 = (unsigned)(unsigned long long)&lA[buf][ar * BK + akq];
    asm volatile("global_load_async_to_lds_b128 %0, %1, off"
                 :: "v"(lo0), "v"(g0) : "memory");
    unsigned long long g1 =
        (unsigned long long)(A + (size_t)(m0 + ar + 64) * K + kt + akq);
    unsigned lo1 = (unsigned)(unsigned long long)&lA[buf][(ar + 64) * BK + akq];
    asm volatile("global_load_async_to_lds_b128 %0, %1, off"
                 :: "v"(lo1), "v"(g1) : "memory");
  };
  auto loadB = [&](int kt) -> uint4 {
    return *((const uint4*)(Bm + (size_t)(kt + bk) * N + n0 + bn));
  };
  auto storeB = [&](uint4 d4, int buf) {
    unsigned short tmp[8];
    *(uint4*)tmp = d4;
#pragma unroll
    for (int j = 0; j < 8; ++j) lB[buf][(bn + j) * BK + bk] = tmp[j];
  };

  // ---- prologue: stage tile 0 ----
  issueA(0, 0);
  storeB(loadB(0), 0);
  asm volatile("s_wait_asynccnt 0x0" ::: "memory");
  __syncthreads();

  int p = 0;
  for (int kt = 0; kt < K; kt += BK) {
    const int nkt = kt + BK;
    const bool has = nkt < K;
    uint4 bnext;
    if (has) {                      // overlap next tile's loads with compute
      issueA(nkt, p ^ 1);
      bnext = loadB(nkt);
    }

    Frag a0, a1, b0, b1;
#pragma unroll
    for (int v = 0; v < 8; ++v) {
      // A 16x32 bf16 layout: koff = (v/4)*16 + half*8 + (v%4)*2
      int ka = ((v >> 2) << 4) + (half << 3) + ((v & 3) << 1);
      // B 32x16 bf16 layout: koff = (v/4)*16 + half*8 + (v%4)*2
      int kb = ((v >> 2) << 4) + (half << 3) + ((v & 3) << 1);
      a0.u[v] = *(const unsigned int*)&lA[p][(wm * 32 + l16) * BK + ka];
      a1.u[v] = *(const unsigned int*)&lA[p][(wm * 32 + 16 + l16) * BK + ka];
      b0.u[v] = *(const unsigned int*)&lB[p][(wn * 32 + l16) * BK + kb];
      b1.u[v] = *(const unsigned int*)&lB[p][(wn * 32 + 16 + l16) * BK + kb];
    }
    acc00 = __builtin_amdgcn_wmma_f32_16x16x32_bf16(false, a0.v, false, b0.v,
                                                    (short)0, acc00, false, false);
    acc01 = __builtin_amdgcn_wmma_f32_16x16x32_bf16(false, a0.v, false, b1.v,
                                                    (short)0, acc01, false, false);
    acc10 = __builtin_amdgcn_wmma_f32_16x16x32_bf16(false, a1.v, false, b0.v,
                                                    (short)0, acc10, false, false);
    acc11 = __builtin_amdgcn_wmma_f32_16x16x32_bf16(false, a1.v, false, b1.v,
                                                    (short)0, acc11, false, false);

    if (has) {
      storeB(bnext, p ^ 1);         // other buffer: no barrier needed before
      asm volatile("s_wait_asynccnt 0x0" ::: "memory");
      __syncthreads();              // single barrier per K-step
      p ^= 1;
    }
  }

#pragma unroll
  for (int r = 0; r < 8; ++r) {
    int row0 = m0 + wm * 32 + r + half * 8;   // C layout: VGPR r -> M=r (+8 hi half)
    int row1 = row0 + 16;
    int c0 = n0 + wn * 32 + l16;
    int c1 = c0 + 16;
    float bi0 = bias ? bias[c0] : 0.f;
    float bi1 = bias ? bias[c1] : 0.f;
    C[(size_t)row0 * N + c0] = acc00[r] + bi0;
    C[(size_t)row0 * N + c1] = acc01[r] + bi1;
    C[(size_t)row1 * N + c0] = acc10[r] + bi0;
    C[(size_t)row1 * N + c1] = acc11[r] + bi1;
  }
}

// ---------------- top-k mask (null score uses only bk) ----------------
// q: [U,B,128] f32; bkvec: [128]; mask: [B,U]
__global__ __launch_bounds__(256) void k_mask(const float* __restrict__ q,
                                              const float* __restrict__ bkvec,
                                              float* __restrict__ mask) {
  int b = blockIdx.x * 256 + threadIdx.x;
  if (b >= Bsz) return;
  float ns[Uu];
  for (int u = 0; u < Uu; ++u) {
    float s = 0.f;
    const float* qp = q + ((size_t)u * Bsz + b) * (HIN_ * QS_);
    for (int d = 0; d < HIN_ * QS_; ++d) s += qp[d] * bkvec[d];
    // -(mean over heads of dot64/sqrt(64)) == -dot128/(HIN*8)
    ns[u] = -s * (1.0f / (HIN_ * 8.0f));
  }
  for (int u = 0; u < Uu; ++u) {
    int rank = 0;
    for (int v = 0; v < Uu; ++v)
      if (ns[v] > ns[u] || (ns[v] == ns[u] && v < u)) ++rank;
    mask[(size_t)b * Uu + u] = (rank < Kk) ? 1.0f : 0.0f;
  }
}

// inp[u,b,s] = mask[b,u] * v0[b,s]  (bf16 out)
__global__ __launch_bounds__(256) void k_make_inp(const float* __restrict__ v0,
                                                  const float* __restrict__ mask,
                                                  unsigned short* __restrict__ inp) {
  int i = blockIdx.x * 256 + threadIdx.x;
  if (i >= Uu * Bsz * VS_) return;
  int u = i / (Bsz * VS_);
  int rem = i - u * (Bsz * VS_);
  int b = rem / VS_, s = rem - b * VS_;
  inp[i] = f2bf(mask[(size_t)b * Uu + u] * v0[(size_t)b * VS_ + s]);
}

// GRU gates for one unit: gx,gh: [B,1536] (biases already added in GEMM epilogue)
__global__ __launch_bounds__(256) void k_gru(const float* __restrict__ gx,
                                             const float* __restrict__ gh,
                                             const float* __restrict__ hsu,
                                             float* __restrict__ hout,
                                             unsigned short* __restrict__ hbf) {
  int i = blockIdx.x * 256 + threadIdx.x;
  if (i >= Bsz * Hh) return;
  int b = i / Hh, j = i - b * Hh;
  size_t base = (size_t)b * G3H + j;
  float rx = gx[base],            rh = gh[base];
  float zx = gx[base + Hh],       zh = gh[base + Hh];
  float nx = gx[base + 2 * Hh],   nh = gh[base + 2 * Hh];
  float r = 1.0f / (1.0f + expf(-(rx + rh)));
  float z = 1.0f / (1.0f + expf(-(zx + zh)));
  float n = tanhf(nx + r * nh);
  float h = (1.0f - z) * n + z * hsu[i];
  hout[i] = h;
  hbf[i] = f2bf(h);
}

// per-b: 6x6 attention + softmax*mask + ctx, out += in place (d_out holds h)
__global__ __launch_bounds__(256) void k_attn(const float* __restrict__ qc,
                                              const float* __restrict__ kc,
                                              const float* __restrict__ vc,
                                              const float* __restrict__ mask,
                                              float* __restrict__ out) {
  __shared__ float att[Uu * Uu];
  __shared__ float probs[Uu * Uu];
  const int b = blockIdx.x;
  const int tid = threadIdx.x;
  const float scale = 1.0f / (HC_ * 5.656854249492381f);  // 1/(HC*sqrt(KC))

  if (tid < Uu * Uu) {
    int u = tid / Uu, v = tid % Uu;
    const float* qp = qc + ((size_t)u * Bsz + b) * (HC_ * QC_);
    const float* kp = kc + ((size_t)v * Bsz + b) * (HC_ * KC_);
    float s = 0.f;
    for (int d = 0; d < HC_ * QC_; ++d) s += qp[d] * kp[d];
    att[tid] = s * scale;
  }
  __syncthreads();
  if (tid < Uu) {
    float m = att[tid * Uu];
    for (int v = 1; v < Uu; ++v) m = fmaxf(m, att[tid * Uu + v]);
    float sum = 0.f, e[Uu];
    for (int v = 0; v < Uu; ++v) { e[v] = expf(att[tid * Uu + v] - m); sum += e[v]; }
    float mk = mask[(size_t)b * Uu + tid];
    for (int v = 0; v < Uu; ++v) probs[tid * Uu + v] = (e[v] / sum) * mk;
  }
  __syncthreads();
  for (int i = tid; i < Uu * Hh; i += 256) {
    int u = i / Hh, d = i - u * Hh;
    size_t oidx = ((size_t)u * Bsz + b) * Hh + d;
    float acc = out[oidx];  // h
    for (int v = 0; v < Uu; ++v)
      acc += probs[u * Uu + v] * vc[((size_t)v * Bsz + b) * Hh + d];
    out[oidx] = acc;
  }
}

// ---------------- host ----------------

extern "C" void kernel_launch(void* const* d_in, const int* in_sizes, int n_in,
                              void* d_out, int out_size, void* d_ws, size_t ws_size,
                              hipStream_t stream) {
  const float* x    = (const float*)d_in[0];
  const float* hs   = (const float*)d_in[1];
  const float* bk   = (const float*)d_in[3];
  const float* Wv   = (const float*)d_in[4];
  const float* bv   = (const float*)d_in[5];
  const float* Wq   = (const float*)d_in[6];
  const float* bq   = (const float*)d_in[7];
  const float* W_ih = (const float*)d_in[8];
  const float* W_hh = (const float*)d_in[9];
  const float* b_ih = (const float*)d_in[10];
  const float* b_hh = (const float*)d_in[11];
  const float* Wqc  = (const float*)d_in[12];
  const float* bqc  = (const float*)d_in[13];
  const float* Wkc  = (const float*)d_in[14];
  const float* Wvc  = (const float*)d_in[16];
  const float* bvc  = (const float*)d_in[17];
  float* out = (float*)d_out;
  (void)in_sizes; (void)n_in; (void)out_size; (void)ws_size;

  char* ws = (char*)d_ws;
  size_t off = 0;
  auto alloc = [&](size_t bytes) -> char* {
    char* p = ws + off;
    off = (off + bytes + 255) & ~(size_t)255;
    return p;
  };

  unsigned short* x_bf   = (unsigned short*)alloc((size_t)Bsz * Dd * 2);
  unsigned short* hs_bf  = (unsigned short*)alloc((size_t)Uu * Bsz * Hh * 2);
  unsigned short* Wq_bf  = (unsigned short*)alloc((size_t)Uu * Hh * 128 * 2);
  unsigned short* Wqc_bf = (unsigned short*)alloc((size_t)Uu * Hh * 128 * 2);
  unsigned short* Wkc_bf = (unsigned short*)alloc((size_t)Uu * Hh * 128 * 2);
  unsigned short* WihT   = (unsigned short*)alloc((size_t)Uu * Dd * G3H * 2);
  unsigned short* WhhT   = (unsigned short*)alloc((size_t)Uu * Hh * G3H * 2);
  unsigned short* Wvm    = (unsigned short*)alloc((size_t)Dd * VS_ * 2);
  float*          bvm    = (float*)alloc((size_t)VS_ * 4);
  unsigned short* Wvcm   = (unsigned short*)alloc((size_t)Uu * Hh * VC_ * 2);
  float*          bvcm   = (float*)alloc((size_t)Uu * VC_ * 4);
  float*          v0     = (float*)alloc((size_t)Bsz * VS_ * 4);
  float*          qbuf   = (float*)alloc((size_t)Uu * Bsz * 128 * 4);
  float*          maskb  = (float*)alloc((size_t)Bsz * Uu * 4);
  unsigned short* inp_bf = (unsigned short*)alloc((size_t)Uu * Bsz * VS_ * 2);
  float*          gxbuf  = (float*)alloc((size_t)Bsz * G3H * 4);
  float*          ghbuf  = (float*)alloc((size_t)Bsz * G3H * 4);
  unsigned short* h_bf   = (unsigned short*)alloc((size_t)Uu * Bsz * Hh * 2);
  float*          qcbuf  = (float*)alloc((size_t)Uu * Bsz * 128 * 4);
  float*          kcbuf  = (float*)alloc((size_t)Uu * Bsz * 128 * 4);
  float*          vcbuf  = (float*)alloc((size_t)Uu * Bsz * VC_ * 4);

  // --- weight / activation conversion ---
  int n;
  n = Bsz * Dd;            k_cvt<<<(n + 255) / 256, 256, 0, stream>>>(x, x_bf, n);
  n = Uu * Bsz * Hh;       k_cvt<<<(n + 255) / 256, 256, 0, stream>>>(hs, hs_bf, n);
  n = Uu * Hh * 128;       k_cvt<<<(n + 255) / 256, 256, 0, stream>>>(Wq, Wq_bf, n);
  n = Uu * Hh * 128;       k_cvt<<<(n + 255) / 256, 256, 0, stream>>>(Wqc, Wqc_bf, n);
  n = Uu * Hh * 128;       k_cvt<<<(n + 255) / 256, 256, 0, stream>>>(Wkc, Wkc_bf, n);
  k_transpose<<<dim3((G3H * Dd + 255) / 256, 1, Uu), 256, 0, stream>>>(W_ih, WihT, G3H, Dd);
  k_transpose<<<dim3((G3H * Hh + 255) / 256, 1, Uu), 256, 0, stream>>>(W_hh, WhhT, G3H, Hh);
  k_headmean<<<dim3((Dd * VS_ + 255) / 256, 1, 1),  256, 0, stream>>>(Wv, bv, Wvm, bvm, Dd, HIN_, VS_);
  k_headmean<<<dim3((Hh * VC_ + 255) / 256, 1, Uu), 256, 0, stream>>>(Wvc, bvc, Wvcm, bvcm, Hh, HC_, VC_);

  // --- v0 = x @ Wv_mean + bv_mean : [B,512] ---
  k_gemm<<<dim3(VS_ / BN, Bsz / BM, 1), 256, 0, stream>>>(
      x_bf, 0, Wvm, 0, bvm, 0, v0, 0, Bsz, VS_, Dd);

  // --- q[u] = hs[u] @ Wq[u] + bq[u] : [U,B,128] ---
  k_gemm<<<dim3(128 / BN, Bsz / BM, Uu), 256, 0, stream>>>(
      hs_bf, (long long)Bsz * Hh, Wq_bf, (long long)Hh * 128,
      bq, 128, qbuf, (long long)Bsz * 128, Bsz, 128, Hh);

  // --- top-k mask + masked input ---
  k_mask<<<(Bsz + 255) / 256, 256, 0, stream>>>(qbuf, bk, maskb);
  n = Uu * Bsz * VS_;
  k_make_inp<<<(n + 255) / 256, 256, 0, stream>>>(v0, maskb, inp_bf);

  // --- per-unit GRU: gx = inp@W_ih^T + b_ih ; gh = hs@W_hh^T + b_hh ---
  for (int u = 0; u < Uu; ++u) {
    k_gemm<<<dim3(G3H / BN, Bsz / BM, 1), 256, 0, stream>>>(
        inp_bf + (size_t)u * Bsz * VS_, 0, WihT + (size_t)u * Dd * G3H, 0,
        b_ih + (size_t)u * G3H, 0, gxbuf, 0, Bsz, G3H, VS_);
    k_gemm<<<dim3(G3H / BN, Bsz / BM, 1), 256, 0, stream>>>(
        hs_bf + (size_t)u * Bsz * Hh, 0, WhhT + (size_t)u * Hh * G3H, 0,
        b_hh + (size_t)u * G3H, 0, ghbuf, 0, Bsz, G3H, Hh);
    n = Bsz * Hh;
    k_gru<<<(n + 255) / 256, 256, 0, stream>>>(
        gxbuf, ghbuf, hs + (size_t)u * Bsz * Hh,
        out + (size_t)u * Bsz * Hh, h_bf + (size_t)u * Bsz * Hh);
  }

  // --- communication attention projections ---
  k_gemm<<<dim3(128 / BN, Bsz / BM, Uu), 256, 0, stream>>>(
      h_bf, (long long)Bsz * Hh, Wqc_bf, (long long)Hh * 128,
      bqc, 128, qcbuf, (long long)Bsz * 128, Bsz, 128, Hh);
  k_gemm<<<dim3(128 / BN, Bsz / BM, Uu), 256, 0, stream>>>(
      h_bf, (long long)Bsz * Hh, Wkc_bf, (long long)Hh * 128,
      (const float*)d_in[15], 128, kcbuf, (long long)Bsz * 128, Bsz, 128, Hh);
  k_gemm<<<dim3(VC_ / BN, Bsz / BM, Uu), 256, 0, stream>>>(
      h_bf, (long long)Bsz * Hh, Wvcm, (long long)Hh * VC_,
      bvcm, VC_, vcbuf, (long long)Bsz * VC_, Bsz, VC_, Hh);

  // --- softmax attention + ctx + h (in place on d_out) ---
  k_attn<<<Bsz, 256, 0, stream>>>(qcbuf, kcbuf, vcbuf, maskb, out);
}
